// CosineRefiner_23983097381174
// MI455X (gfx1250) — matrix operand order repaired and checked by reading, not compiled, not run on hardware
//
#include <hip/hip_runtime.h>
#include <hip/hip_bf16.h>
#include <math.h>

// ---------------------------------------------------------------------------
// CosineRefiner for MI455X (gfx1250, wave32).
// fL/fR: (4,256,96,320) f32, disp_full: (4,1,384,1280) f32.
// Correlation band computed via v_wmma_f32_16x16x32_f16 (f32 accumulate),
// normalization folded in as a post-scale (sum-of-squares from same loads).
// Cache policy: fL / disp_full are single-use streams -> NT loads; final
// outputs are write-once -> NT stores; fR window (2x reuse) and the small
// inter-kernel intermediates keep default RT so they stay L2-resident.
// ---------------------------------------------------------------------------

typedef __attribute__((ext_vector_type(16))) _Float16 v16h;
typedef __attribute__((ext_vector_type(8)))  float    v8f;

#define Wc   320
#define Hc   96
#define Cc   256
#define Bc   4
#define HWs  (Hc * Wc)          // 30720
#define CHW  (Cc * HWs)         // 7864320
#define WF   1280
#define HF   384
#define HWF  (HF * WF)          // 491520

// ------------------------- kernel 1: correlation ---------------------------
// grid = (20, 96, 4), block = 32 (one wave per 16-pixel x-tile).
// A (WMMA SRC0) = fR window rows M=0..15 -> x' = x0-8+M (block0) / x0+8+M (block1)
// B (WMMA SRC1) = fL tile columns N=0..15 -> x = x0+N
// A layout (16b 16x32): lane L<16 holds M=L, Klocal {0..7,16..23}; lane L+16: {8..15,24..31}
// B layout (16b 32x16): lane L<16 holds N=L, Klocal {0..7,16..23}; lane L+16: {8..15,24..31}
// C/D layout: lane L<16 holds N=L, M=j (VGPR j); lane L+16 holds N=L-16, M=j+8.
__global__ __launch_bounds__(32)
void corr_kernel(const float* __restrict__ fL, const float* __restrict__ fR,
                 const float* __restrict__ dispFull,
                 float* __restrict__ dispRef, float* __restrict__ confOut)
{
    const int x0   = blockIdx.x * 16;
    const int y    = blockIdx.y;
    const int b    = blockIdx.z;
    const int lane = threadIdx.x;
    const int n    = lane & 15;
    const int half = lane >> 4;
    const int kbA  = half * 8;      // A K-base within 32-chunk
    const int kbB  = half * 16;     // unused; B indexes with kbA

    const float* fLrow = fL + b * CHW + y * Wc;
    const float* fRrow = fR + b * CHW + y * Wc;

    const int xl  = x0 + n;                       // left pixel (never wraps)
    const int xw0 = (x0 - 8 + n + Wc) % Wc;       // right window, block 0 (roll wrap)
    const int xw1 = (x0 + 8 + n) % Wc;            // right window, block 1

    v8f d0 = {}; v8f d1 = {};
    float ssA0 = 0.f, ssA1 = 0.f, ssB = 0.f;      // per-lane partial sum-of-squares

    for (int kc = 0; kc < 8; ++kc) {
        const int cb = kc * 32;
        v16h a0, a1, bm;
#pragma unroll
        for (int i = 0; i < 16; ++i) {
            const int cA = cb + kbA + ((i < 8) ? i : i + 8);  // {0..7,16..23} (+half*8)
            const int cB = cb + kbA + ((i < 8) ? i : i + 8);  // {0..7,16..23} (+half*8)
            const float r0 = fRrow[cA * HWs + xw0];           // fR: L2 reuse -> RT
            const float r1 = fRrow[cA * HWs + xw1];
            const float l  = __builtin_nontemporal_load(&fLrow[cB * HWs + xl]); // single-use
            ssA0 = fmaf(r0, r0, ssA0);
            ssA1 = fmaf(r1, r1, ssA1);
            ssB  = fmaf(l,  l,  ssB);
            a0[i] = (_Float16)r0;
            a1[i] = (_Float16)r1;
            bm[i] = (_Float16)l;
        }
        d0 = __builtin_amdgcn_wmma_f32_16x16x32_f16(false, a0, false, bm,
                                                    (short)0, d0, false, false);
        d1 = __builtin_amdgcn_wmma_f32_16x16x32_f16(false, a1, false, bm,
                                                    (short)0, d1, false, false);
    }

    // Full norms: lane n & lane n+16 hold complementary K-halves of the same row/col.
    __shared__ float ldsInvR[32];
    __shared__ float ldsInvB[16];
    ssA0 += __shfl_xor(ssA0, 16, 32);
    ssA1 += __shfl_xor(ssA1, 16, 32);
    ssB  += __shfl_xor(ssB,  16, 32);
    if (half == 0) {
        ldsInvR[n]      = 1.0f / fmaxf(sqrtf(ssA0), 1e-6f);   // x' = x0-8+n
        ldsInvR[n + 16] = 1.0f / fmaxf(sqrtf(ssA1), 1e-6f);   // x' = x0+8+n
        ldsInvB[n]      = 1.0f / fmaxf(sqrtf(ssB),  1e-6f);   // x  = x0+n
    }
    __syncthreads();

    // Scale accumulators into cosine sims. This lane holds, for left pixel x0+n,
    // sims at M_global = kbA+j (block0) and 16+kbA+j (block1), M_global = x'-(x0-8).
    const float invB = ldsInvB[n];
    float s0[8], s1[8];
#pragma unroll
    for (int j = 0; j < 8; ++j) {
        s0[j] = d0[j] * ldsInvR[kbA + j]      * invB;
        s1[j] = d1[j] * ldsInvR[16 + kbA + j] * invB;
    }

    // Offsets d = n+8-M_global; valid window d in [-3,3] <=> M_global in [n+5,n+11].
    const int lo = n + 5, hi = n + 11;
    float m = -3.0e38f;
#pragma unroll
    for (int j = 0; j < 8; ++j) {
        const int M0 = kbA + j;      if (M0 >= lo && M0 <= hi) m = fmaxf(m, s0[j]);
        const int M1 = 16 + kbA + j; if (M1 >= lo && M1 <= hi) m = fmaxf(m, s1[j]);
    }
    m = fmaxf(m, __shfl_xor(m, 16, 32));
    float es = 0.f, ed = 0.f;
#pragma unroll
    for (int j = 0; j < 8; ++j) {
        const int M0 = kbA + j;
        if (M0 >= lo && M0 <= hi) {
            const float e = __expf((s0[j] - m) * 10.0f);      // 1/TEMP = 10
            es += e; ed += e * (float)(n + 8 - M0);
        }
        const int M1 = 16 + kbA + j;
        if (M1 >= lo && M1 <= hi) {
            const float e = __expf((s1[j] - m) * 10.0f);
            es += e; ed += e * (float)(n + 8 - M1);
        }
    }
    es += __shfl_xor(es, 16, 32);
    ed += __shfl_xor(ed, 16, 32);
    const float delta = ed / es;
    const float conf  = 1.0f / es;   // p_max = exp(0)/es since m is the global max

    if (half == 0) {
        // disp0: 4x bilinear downsample degenerates to the avg of the 2x2 block
        // at rows 4y+1..4y+2, cols 4x+1..4x+2; then /4 disparity scale => *0.0625.
        const int x = x0 + n;
        const float* df = dispFull + b * HWF;
        const int ry = 4 * y + 1, rx = 4 * x + 1;
        const float sum = __builtin_nontemporal_load(&df[ry * WF + rx])
                        + __builtin_nontemporal_load(&df[ry * WF + rx + 1])
                        + __builtin_nontemporal_load(&df[(ry + 1) * WF + rx])
                        + __builtin_nontemporal_load(&df[(ry + 1) * WF + rx + 1]);
        const float disp0 = sum * 0.0625f;
        const int o = b * HWs + y * Wc + x;
        dispRef[o] = disp0 + delta;     // RT store: consumed by lap_kernel from L2
        confOut[o] = conf;              // RT store: consumed by up_kernel from L2
    }
}

// ------------------------- kernel 2: laplacian sharpen ---------------------
__global__ __launch_bounds__(256)
void lap_kernel(const float* __restrict__ dispRef, float* __restrict__ dispQ)
{
    const int idx = blockIdx.x * 256 + threadIdx.x;     // < 4*96*320
    const int x = idx % Wc;
    const int y = (idx / Wc) % Hc;
    const float c = dispRef[idx];
    const float up    = (y > 0)      ? dispRef[idx - Wc] : 0.f;
    const float down  = (y < Hc - 1) ? dispRef[idx + Wc] : 0.f;
    const float left  = (x > 0)      ? dispRef[idx - 1]  : 0.f;
    const float right = (x < Wc - 1) ? dispRef[idx + 1]  : 0.f;
    const float lap = up + down + left + right - 4.f * c;
    dispQ[idx] = c - 0.2f * lap;
}

// ------------------------- kernel 3: 4x bilinear upsample ------------------
__global__ __launch_bounds__(256)
void up_kernel(const float* __restrict__ dispQ, const float* __restrict__ conf,
               float* __restrict__ outD, float* __restrict__ outC)
{
    const int idx = blockIdx.x * 256 + threadIdx.x;     // < 4*384*1280
    const int X = idx % WF;
    const int t = idx / WF;
    const int Y = t % HF;
    const int b = t / HF;

    const float ys = fmaxf(0.25f * (float)Y - 0.375f, 0.0f);
    const float xs = fmaxf(0.25f * (float)X - 0.375f, 0.0f);
    const int y0 = (int)ys;                 // ys >= 0, floor == trunc
    const int x0 = (int)xs;
    const float wy = ys - (float)y0;
    const float wx = xs - (float)x0;
    const int y1 = min(y0 + 1, Hc - 1);
    const int x1 = min(x0 + 1, Wc - 1);

    const int base = b * HWs;
    const int i00 = base + y0 * Wc + x0, i01 = base + y0 * Wc + x1;
    const int i10 = base + y1 * Wc + x0, i11 = base + y1 * Wc + x1;
    const float w00 = (1.f - wy) * (1.f - wx), w01 = (1.f - wy) * wx;
    const float w10 = wy * (1.f - wx),         w11 = wy * wx;

    const float dv = w00 * dispQ[i00] + w01 * dispQ[i01]
                   + w10 * dispQ[i10] + w11 * dispQ[i11];
    const float cv = w00 * conf[i00] + w01 * conf[i01]
                   + w10 * conf[i10] + w11 * conf[i11];

    // Write-once outputs: NT stores, don't displace L2-resident working data.
    __builtin_nontemporal_store(dv * 4.0f, &outD[idx]);
    __builtin_nontemporal_store(cv,        &outC[idx]);
}

// ---------------------------------------------------------------------------
extern "C" void kernel_launch(void* const* d_in, const int* in_sizes, int n_in,
                              void* d_out, int out_size, void* d_ws, size_t ws_size,
                              hipStream_t stream)
{
    const float* fL       = (const float*)d_in[0];
    const float* fR       = (const float*)d_in[1];
    const float* dispFull = (const float*)d_in[2];

    float* ws      = (float*)d_ws;
    float* dispRef = ws;                        // 4*96*320
    float* confBuf = ws + Bc * HWs;             // 4*96*320
    float* dispQ   = ws + 2 * Bc * HWs;         // 4*96*320

    float* outD = (float*)d_out;                // 4*384*1280
    float* outC = outD + Bc * HWF;              // 4*384*1280

    dim3 g1(Wc / 16, Hc, Bc);                   // (20, 96, 4)
    corr_kernel<<<g1, 32, 0, stream>>>(fL, fR, dispFull, dispRef, confBuf);

    lap_kernel<<<(Bc * HWs) / 256, 256, 0, stream>>>(dispRef, dispQ);

    up_kernel<<<(Bc * HWF) / 256, 256, 0, stream>>>(dispQ, confBuf, outD, outC);
}
